// SIMCLR_39900246180360
// MI455X (gfx1250) — matrix operand order repaired and checked by reading, not compiled, not run on hardware
//
#include <hip/hip_runtime.h>
#include <hip/hip_bf16.h>
#include <math.h>

typedef _Float16 f16;
typedef __attribute__((ext_vector_type(16))) _Float16 v16h;
typedef __attribute__((ext_vector_type(8)))  _Float16 v8h;
typedef __attribute__((ext_vector_type(8)))  float    v8f;

#define N_ROWS 4096   // per view
#define ZDIM   2048
#define TOTAL  8192   // 2N

// ---------------------------------------------------------------------------
// Kernel 1: row-normalize z1,z2 (fp32 math), write f16 reps R[8192][2048],
// positive logits pos[4096] = (a.b)/T, and zero rowsum[8192].
// One block (256 threads) per row pair; each thread owns 8 strided elements.
// ---------------------------------------------------------------------------
__global__ void __launch_bounds__(256)
simclr_normalize(const float* __restrict__ z1, const float* __restrict__ z2,
                 f16* __restrict__ R, float* __restrict__ pos,
                 float* __restrict__ rowsum) {
    const int row = blockIdx.x;
    const int t   = threadIdx.x;
    const float* p1 = z1 + (size_t)row * ZDIM;
    const float* p2 = z2 + (size_t)row * ZDIM;

    float x1[8], x2[8];
    float s1 = 0.f, s2 = 0.f, sd = 0.f;
#pragma unroll
    for (int j = 0; j < 8; ++j) {
        float a = p1[t + 256 * j];
        float b = p2[t + 256 * j];
        x1[j] = a; x2[j] = b;
        s1 += a * a; s2 += b * b; sd += a * b;
    }
    // wave32 reduction then cross-wave via LDS
#pragma unroll
    for (int m = 16; m > 0; m >>= 1) {
        s1 += __shfl_xor(s1, m, 32);
        s2 += __shfl_xor(s2, m, 32);
        sd += __shfl_xor(sd, m, 32);
    }
    __shared__ float sh[3][8];
    const int wave = t >> 5, lane = t & 31;
    if (lane == 0) { sh[0][wave] = s1; sh[1][wave] = s2; sh[2][wave] = sd; }
    __syncthreads();
    float t1 = 0.f, t2 = 0.f, td = 0.f;
#pragma unroll
    for (int w = 0; w < 8; ++w) { t1 += sh[0][w]; t2 += sh[1][w]; td += sh[2][w]; }

    const float r1 = 1.0f / sqrtf(t1);
    const float r2 = 1.0f / sqrtf(t2);

    f16* o1 = R + (size_t)row * ZDIM;
    f16* o2 = R + (size_t)(row + N_ROWS) * ZDIM;
#pragma unroll
    for (int j = 0; j < 8; ++j) {
        o1[t + 256 * j] = (f16)(x1[j] * r1);
        o2[t + 256 * j] = (f16)(x2[j] * r2);
    }
    if (t == 0) {
        pos[row] = td * r1 * r2 * 2.0f;      // /T, T = 0.5
        rowsum[row]          = 0.f;
        rowsum[row + N_ROWS] = 0.f;
    }
}

// ---------------------------------------------------------------------------
// Kernel 2: symmetric Gram + exp-sum, upper-triangular tiles only (2x fewer
// WMMAs than the full matrix). Each block owns one 128x128 tile (I,J), I<=J:
//   - tile enumeration: grid(64,33), (bx,(bx+by)%64) circular pairing; offset
//     32 is double-covered so by==32 && bx>=32 exits (uniform, pre-WMMA).
//   - block = 8 waves; wave w computes rows [I+16w,+16) x cols [J,+128) with
//     8 v_wmma_f32_16x16x32_f16 accumulators over K=2048.
//   - e = exp(2*dot-2) folds into row sums (rows I..) and, for off-diagonal
//     tiles, column sums (rows J.. by symmetry s_ij == s_ji) -> atomicAdd.
// Fragments load straight from L2 (32MB of reps << 192MB L2):
//   A frag lane(l,h): row I+16w+l, K chunks [kb+16h,+8) & [kb+16h+8,+8) (2x16B)
//   B frag lane(l,h): row J+16n+l, K chunk  [kb+16h,+16)               (1x32B)
// ---------------------------------------------------------------------------
__global__ void __launch_bounds__(256)
simclr_gram_expsum(const f16* __restrict__ R, float* __restrict__ rowsum) {
    const int bx = blockIdx.x;          // 0..63
    const int by = blockIdx.y;          // 0..32
    if (by == 32 && bx >= 32) return;   // de-duplicate circular offset 32
    const int p = bx, q = (bx + by) & 63;
    const int bi = (p < q) ? p : q;
    const int bj = (p < q) ? q : p;
    const int I = bi * 128, J = bj * 128;
    const bool offdiag = (bi != bj);

    const int wave = threadIdx.x >> 5;
    const int lane = threadIdx.x & 31;
    const int l = lane & 15;
    const int h = lane >> 4;
    const int rowBase = I + wave * 16;
    const size_t arow = (size_t)(rowBase + l) * ZDIM;

    v8f acc[8] = {};
    for (int kt = 0; kt < 64; ++kt) {
        const int kb = kt * 32;
        v16h A;
        v8h* ap = reinterpret_cast<v8h*>(&A);
        ap[0] = *reinterpret_cast<const v8h*>(R + arow + kb + h * 16);
        ap[1] = *reinterpret_cast<const v8h*>(R + arow + kb + h * 16 + 8);
#pragma unroll
        for (int n = 0; n < 8; ++n) {
            const f16* bp = R + (size_t)(J + n * 16 + l) * ZDIM + kb + h * 16;
            v16h B = *reinterpret_cast<const v16h*>(bp);
            acc[n] = __builtin_amdgcn_wmma_f32_16x16x32_f16(
                /*neg_a=*/false, A, /*neg_b=*/false, B,
                /*c_mod=*/(short)0, acc[n],
                /*reuse_a=*/false, /*reuse_b=*/false);
        }
    }

    // C/D layout: VGPR r, lanes 0-15 -> row M=r ; lanes 16-31 -> row M=r+8;
    // column N = l within each half. term = exp(logit - 2), logit = dot/0.5.
    float rs[8] = {0.f, 0.f, 0.f, 0.f, 0.f, 0.f, 0.f, 0.f};
#pragma unroll
    for (int n = 0; n < 8; ++n) {
        float cs = 0.f;
#pragma unroll
        for (int r = 0; r < 8; ++r) {
            float e = __expf(2.0f * acc[n][r] - 2.0f);
            rs[r] += e;
            cs += e;
        }
        if (offdiag) {
            // column sum over this wave's 16 rows -> row (J + n*16 + l)'s sum
            cs += __shfl_xor(cs, 16, 32);
            if (h == 0) atomicAdd(&rowsum[J + n * 16 + l], cs);
        }
    }
    // row sums: reduce over the 16 column-lanes of each half
#pragma unroll
    for (int r = 0; r < 8; ++r) {
        float v = rs[r];
        v += __shfl_xor(v, 1, 32);
        v += __shfl_xor(v, 2, 32);
        v += __shfl_xor(v, 4, 32);
        v += __shfl_xor(v, 8, 32);
        if (l == 0) atomicAdd(&rowsum[rowBase + r + 8 * h], v);
    }
}

// ---------------------------------------------------------------------------
// Kernel 3: loss = mean_i( log(rowsum[i] - 1) + 2 - pos[i & 4095] )
// (the -1 removes the diagonal term exp(s_ii - 2) ~= 1)
// ---------------------------------------------------------------------------
__global__ void __launch_bounds__(256)
simclr_finalize(const float* __restrict__ rowsum, const float* __restrict__ pos,
                float* __restrict__ out) {
    __shared__ float sh[256];
    const int t = threadIdx.x;
    float s = 0.f;
    for (int i = t; i < TOTAL; i += 256)
        s += logf(rowsum[i] - 1.0f) + 2.0f - pos[i & (N_ROWS - 1)];
    sh[t] = s;
    __syncthreads();
    for (int off = 128; off > 0; off >>= 1) {
        if (t < off) sh[t] += sh[t + off];
        __syncthreads();
    }
    if (t == 0) out[0] = sh[0] * (1.0f / (float)TOTAL);
}

extern "C" void kernel_launch(void* const* d_in, const int* in_sizes, int n_in,
                              void* d_out, int out_size, void* d_ws, size_t ws_size,
                              hipStream_t stream) {
    (void)in_sizes; (void)n_in; (void)out_size; (void)ws_size;
    const float* z1 = (const float*)d_in[0];
    const float* z2 = (const float*)d_in[1];
    float* out = (float*)d_out;

    char* ws = (char*)d_ws;
    f16*   R      = (f16*)ws;                                    // 32 MB
    float* rowsum = (float*)(ws + (size_t)TOTAL * ZDIM * sizeof(f16));
    float* pos    = rowsum + TOTAL;

    simclr_normalize<<<N_ROWS, 256, 0, stream>>>(z1, z2, R, pos, rowsum);
    dim3 grid(TOTAL / 128, 33);   // upper-triangular tile pairing
    simclr_gram_expsum<<<grid, 256, 0, stream>>>(R, rowsum);
    simclr_finalize<<<1, 256, 0, stream>>>(rowsum, pos, out);
}
